// BERT_tensor_2714419331578
// MI455X (gfx1250) — matrix operand order repaired and checked by reading, not compiled, not run on hardware
//
#include <hip/hip_runtime.h>

// ---------------------------------------------------------------------------
// BERT with tensor-network heads, forward pass, for gfx1250 (MI455X, wave32).
// All GEMMs use v_wmma_f32_16x16x32_bf16 with f32 accumulation, and all take
// the NT path (B stored [N][K] row-major) so every LDS store is b128.
// B=32, S=200 (padded to SP=256 for attention), D=256, H=6, DFF=1024, L=8.
// Fully tile-aligned (M,N % 64 == 0, K % 32 == 0): guard-free b128 loaders,
// double-buffered LDS, one barrier per K-step; only epilogue stores predicated.
// ---------------------------------------------------------------------------

typedef __attribute__((ext_vector_type(16))) __bf16 v16bf;
typedef __attribute__((ext_vector_type(8)))  __bf16 v8bf;
typedef __attribute__((ext_vector_type(8)))  float  v8f;

#define NB   32
#define NS   200
#define SP   256              // padded sequence length for attention tiles
#define ND   256
#define NH   6
#define NFF  1024
#define NL   8
#define NM   (NB*NS)          // 6400 rows

// ---------------------------------------------------------------------------
// Batched bf16 WMMA GEMM (NT):  C[z] = A[z] (MxK) * B[z]^T, B stored [N][K].
// Requires grid.y*64 rows / grid.x*64 cols in the buffers, K % 32 == 0.
// Stores predicated on row < Mstore only.
// Per-z offsets: off = (z/zdiv)*s?b + (z%zdiv)*s?h (elements).
// Block: 256 thr = 8 waves; tile 64x64; wave grid 4x2, wave tile 16x32.
// LDS: A M-major, B N-major, both [2][64][40] (double buffered, 20 KB);
// every lane fragment is two contiguous 16B runs -> ds_load_b128.
// ---------------------------------------------------------------------------
__global__ void gemm_bf16_wmma(const __bf16* __restrict__ A,
                               const __bf16* __restrict__ Bm,
                               const float*  __restrict__ bias,
                               float* __restrict__ Cf, __bf16* __restrict__ Cb,
                               int Mstore, int K, int lda, int ldb, int ldc,
                               int zdiv, long sAb, long sAh, long sBb, long sBh,
                               long sCb, long sCh, long sBiasb, int relu)
{
    __shared__ alignas(16) __bf16 As[2][64][40];   // [buf][m][k], row stride 80 B
    __shared__ alignas(16) __bf16 Bs[2][64][40];   // [buf][n][k], row stride 80 B

    const int tid  = threadIdx.x;
    const int z    = blockIdx.z;
    const long zb  = z / zdiv, zh = z % zdiv;
    const __bf16* Ab = A  + zb*sAb + zh*sAh;
    const __bf16* Bb = Bm + zb*sBb + zh*sBh;
    const float*  bp = bias ? (bias + zb*sBiasb) : nullptr;
    const long coff  = zb*sCb + zh*sCh;

    const int m0 = blockIdx.y * 64, n0 = blockIdx.x * 64;
    const int w = tid >> 5, lane = tid & 31, lm = lane & 15, lh = lane >> 4;
    const int wm = w & 3, wn = w >> 2;

    v8f acc0 = {}; v8f acc1 = {};

    // loaders: 64 rows x (4 x 8 K-elements); one b128 per operand per thread
    const int ar = tid >> 2, ac = (tid & 3) * 8;
    const __bf16* gA = Ab + (long)(m0 + ar)*lda + ac;
    const __bf16* gB = Bb + (long)(n0 + ar)*ldb + ac;

    // prologue: stage tile 0 into buffer 0
    *(v8bf*)&As[0][ar][ac] = *(const v8bf*)gA;
    *(v8bf*)&Bs[0][ar][ac] = *(const v8bf*)gB;

    int cur = 0;
    for (int k0 = 0; k0 < K; k0 += 32) {
        __syncthreads();                 // tile[cur] stores visible; prev reads done
        const int kn = k0 + 32;
        if (kn < K) {                    // stage next tile while computing this one
            *(v8bf*)&As[cur^1][ar][ac] = *(const v8bf*)(gA + kn);
            *(v8bf*)&Bs[cur^1][ar][ac] = *(const v8bf*)(gB + kn);
            if (kn + 32 < K) {
                __builtin_prefetch(gA + kn + 32, 0, 1);
                __builtin_prefetch(gB + kn + 32, 0, 1);
            }
        }

        // A fragment (16-bit A 16x32): lanes 0-15 own K 0-15,
        // lanes 16-31 own K 16-31; row M = lane&15. Two ds_load_b128.
        const int arow = wm*16 + lm, kb = lh*16;
        const v8bf a_lo = *(const v8bf*)&As[cur][arow][kb];
        const v8bf a_hi = *(const v8bf*)&As[cur][arow][kb + 8];
        const v16bf afrag = __builtin_shufflevector(a_lo, a_hi,
            0,1,2,3,4,5,6,7,8,9,10,11,12,13,14,15);

        // B fragment: lanes 0-15 hold K 0-15, lanes 16-31 hold K 16-31; col = lane&15.
        const int kb2 = lh*16, bc0 = wn*32 + lm, bc1 = bc0 + 16;
        const v8bf b0lo = *(const v8bf*)&Bs[cur][bc0][kb2];
        const v8bf b0hi = *(const v8bf*)&Bs[cur][bc0][kb2 + 8];
        const v8bf b1lo = *(const v8bf*)&Bs[cur][bc1][kb2];
        const v8bf b1hi = *(const v8bf*)&Bs[cur][bc1][kb2 + 8];
        const v16bf bfrag0 = __builtin_shufflevector(b0lo, b0hi,
            0,1,2,3,4,5,6,7,8,9,10,11,12,13,14,15);
        const v16bf bfrag1 = __builtin_shufflevector(b1lo, b1hi,
            0,1,2,3,4,5,6,7,8,9,10,11,12,13,14,15);

        acc0 = __builtin_amdgcn_wmma_f32_16x16x32_bf16(false, afrag, false, bfrag0,
                                                       (short)0, acc0, false, false);
        acc1 = __builtin_amdgcn_wmma_f32_16x16x32_bf16(false, afrag, false, bfrag1,
                                                       (short)0, acc1, false, false);
        cur ^= 1;
    }

    // D layout: VGPR r -> M = r + 8*(lane>=16); N = lane&15.
    const int c0 = n0 + wn*32 + lm, c1 = c0 + 16;
    const float bias0 = bp ? bp[c0] : 0.0f;
    const float bias1 = bp ? bp[c1] : 0.0f;
    #pragma unroll
    for (int r = 0; r < 8; ++r) {
        const int row = m0 + wm*16 + lh*8 + r;
        if (row >= Mstore) continue;
        float v0 = acc0[r] + bias0, v1 = acc1[r] + bias1;
        if (relu) { v0 = fmaxf(v0, 0.0f); v1 = fmaxf(v1, 0.0f); }
        const long rb = coff + (long)row*ldc;
        if (Cf) { Cf[rb+c0] = v0;         Cf[rb+c1] = v1; }
        if (Cb) { Cb[rb+c0] = (__bf16)v0; Cb[rb+c1] = (__bf16)v1; }
    }
}

// ---------------------------------------------------------------------------
// Embedding: h[b,s,:] = tok_emb[tokens[b,s],:] + pos_emb[s,:]
// ---------------------------------------------------------------------------
__global__ void embed_kernel(const int* __restrict__ tokens,
                             const float* __restrict__ tok_emb,
                             const float* __restrict__ pos_emb,
                             float* __restrict__ hf, __bf16* __restrict__ hb)
{
    const long i = (long)blockIdx.x * 256 + threadIdx.x;
    if (i >= (long)NM * ND) return;
    const int d = i & (ND - 1);
    const long sf = i >> 8;
    const int s = (int)(sf % NS);
    const int tok = tokens[sf];
    const float v = tok_emb[(long)tok * ND + d] + pos_emb[(long)s * ND + d];
    hf[i] = v; hb[i] = (__bf16)v;
}

// ---------------------------------------------------------------------------
// Materialize MPO weight TRANSPOSED: WT[x][n=(i,j,k,l)][d=(p,q,r,t)] =
//   sum_{m,n,o} A1[x,p,m,i] A2[x,q,m,n,j] A3[x,r,n,o,k] A4[x,t,o,l]   (TD=2)
// ---------------------------------------------------------------------------
__global__ void build_wtns(const float* __restrict__ A1, const float* __restrict__ A2,
                           const float* __restrict__ A3, const float* __restrict__ A4,
                           __bf16* __restrict__ WT)
{
    const int i = blockIdx.x * blockDim.x + threadIdx.x;
    if (i >= 3 * 256 * 1024) return;
    const int n = i & 1023, d = (i >> 10) & 255, x = i >> 18;
    const int p = (d >> 6) & 3, q = (d >> 4) & 3, r = (d >> 2) & 3, t = d & 3;
    const int ii = (n >> 8) & 3, j = (n >> 5) & 7, k = (n >> 2) & 7, l2 = n & 3;
    float s = 0.0f;
    for (int m = 0; m < 2; ++m) {
        const float a1v = A1[((x*4 + p)*2 + m)*4 + ii];
        for (int nn = 0; nn < 2; ++nn) {
            const float a12 = a1v * A2[(((x*4 + q)*2 + m)*2 + nn)*8 + j];
            for (int o = 0; o < 2; ++o)
                s += a12 * A3[(((x*4 + r)*2 + nn)*2 + o)*8 + k]
                         * A4[((x*4 + t)*2 + o)*4 + l2];
        }
    }
    WT[x*262144 + n*256 + d] = (__bf16)s;    // [x][N=1024][K=256]
}

// ---------------------------------------------------------------------------
// Assemble q/k into [x][B][H][SP][256] and V TRANSPOSED into [B][H][256][SP]
// (bf16, zero-filled in the padded s range) from cls [3][M][512] / tns [3][M][1024].
// ---------------------------------------------------------------------------
__global__ void assemble_qkv(const __bf16* __restrict__ cls,
                             const __bf16* __restrict__ tns,
                             __bf16* __restrict__ qkv)
{
    const long i = (long)blockIdx.x * 256 + threadIdx.x;
    if (i >= 3L * NB * NH * SP * ND) return;
    long t1 = i >> 16;                       // 64K elements per (x,b,h) slab
    const int h = (int)(t1 % NH); t1 /= NH;
    const int b = (int)(t1 % NB);
    const int x = (int)(t1 / NB);
    int s, d;
    if (x < 2) { d = (int)(i & (ND - 1)); s = (int)((i >> 8) & (SP - 1)); } // [s][d]
    else       { s = (int)(i & (SP - 1)); d = (int)((i >> 8) & (ND - 1)); } // [d][s]
    __bf16 v = (__bf16)0.0f;
    if (s < NS) {
        const long m = (long)b * NS + s;
        if (h < 2) v = cls[((long)x * NM + m) * 512  + h * ND + d];
        else       v = tns[((long)x * NM + m) * 1024 + (h - 2) * ND + d];
    }
    qkv[i] = v;
}

// ---------------------------------------------------------------------------
// Masked softmax over keys: attn = softmax(scores/16 + (tok==0)*-1e9) -> bf16.
// Writes the full padded row (zeros for cols >= 200) so P*V can run K=SP.
// grid (NS, B*H), block SP
// ---------------------------------------------------------------------------
__global__ void softmax_bias(const float* __restrict__ scores,
                             const int* __restrict__ tokens,
                             __bf16* __restrict__ attn)
{
    const int qrow = blockIdx.x, z = blockIdx.y, b = z / NH;
    const int tid = threadIdx.x;
    __shared__ float red[SP];
    const long base = (long)z * SP * SP + (long)qrow * SP;
    const bool valid = tid < NS;
    float x = -3.0e38f;
    if (valid) {
        x = scores[base + tid] * 0.0625f;
        if (tokens[b * NS + tid] == 0) x += -1.0e9f;
    }
    red[tid] = x; __syncthreads();
    for (int s = SP/2; s > 0; s >>= 1) { if (tid < s) red[tid] = fmaxf(red[tid], red[tid+s]); __syncthreads(); }
    const float mx = red[0]; __syncthreads();
    const float e = valid ? __expf(x - mx) : 0.0f;
    red[tid] = e; __syncthreads();
    for (int s = SP/2; s > 0; s >>= 1) { if (tid < s) red[tid] += red[tid+s]; __syncthreads(); }
    const float inv = 1.0f / red[0];
    attn[base + tid] = (__bf16)(e * inv);     // zero in the pad columns
}

// ---------------------------------------------------------------------------
// out = LayerNorm(x + y) * g + b  (row = blockIdx.x, D=256), f32 + bf16 outputs
// ---------------------------------------------------------------------------
__global__ void add_layernorm(const float* __restrict__ x, const float* __restrict__ y,
                              const float* __restrict__ g, const float* __restrict__ be,
                              float* __restrict__ outf, __bf16* __restrict__ outb)
{
    const int row = blockIdx.x, tid = threadIdx.x;
    __shared__ float red[256];
    const long idx = (long)row * ND + tid;
    const float v = x[idx] + y[idx];
    red[tid] = v; __syncthreads();
    for (int s = 128; s > 0; s >>= 1) { if (tid < s) red[tid] += red[tid+s]; __syncthreads(); }
    const float mean = red[0] * (1.0f / ND); __syncthreads();
    const float dv = v - mean;
    red[tid] = dv * dv; __syncthreads();
    for (int s = 128; s > 0; s >>= 1) { if (tid < s) red[tid] += red[tid+s]; __syncthreads(); }
    const float rs = rsqrtf(red[0] * (1.0f / ND) + 1e-6f);
    const float o = dv * rs * g[tid] + be[tid];
    outf[idx] = o; outb[idx] = (__bf16)o;
}

// ---------------------------------------------------------------------------
// Transposing convert: src [K][N] f32 -> dst [N][K] bf16 (grid.z batches)
// ---------------------------------------------------------------------------
__global__ void f32_to_bf16_t(const float* __restrict__ src, __bf16* __restrict__ dst,
                              int K, int N)
{
    const long base = (long)blockIdx.z * K * N;
    const int i = blockIdx.x * 256 + threadIdx.x;      // i = n*K + k (dst-linear)
    if (i >= K * N) return;
    const int k = i % K, n = i / K;
    dst[base + i] = (__bf16)src[base + (long)k*N + n];
}

__global__ void copy_f32(const float* __restrict__ src, float* __restrict__ dst, long n)
{
    const long i = (long)blockIdx.x * 256 + threadIdx.x;
    if (i < n) dst[i] = src[i];
}

// ---------------------------------------------------------------------------
extern "C" void kernel_launch(void* const* d_in, const int* in_sizes, int n_in,
                              void* d_out, int out_size, void* d_ws, size_t ws_size,
                              hipStream_t stream)
{
    (void)in_sizes; (void)n_in; (void)out_size; (void)ws_size;
    const int*   tokens  = (const int*)  d_in[0];
    const float* tok_emb = (const float*)d_in[1];
    const float* pos_emb = (const float*)d_in[2];
    const float* wqkv_w  = (const float*)d_in[3];
    const float* wqkv_b  = (const float*)d_in[4];
    const float* A1      = (const float*)d_in[5];
    const float* A2      = (const float*)d_in[6];
    const float* A3      = (const float*)d_in[7];
    const float* A4      = (const float*)d_in[8];
    const float* tnb     = (const float*)d_in[9];
    const float* out_w   = (const float*)d_in[10];
    const float* out_b   = (const float*)d_in[11];
    const float* ff1_w   = (const float*)d_in[12];
    const float* ff1_b   = (const float*)d_in[13];
    const float* ff2_w   = (const float*)d_in[14];
    const float* ff2_b   = (const float*)d_in[15];
    const float* ln1_g   = (const float*)d_in[16];
    const float* ln1_b   = (const float*)d_in[17];
    const float* ln2_g   = (const float*)d_in[18];
    const float* ln2_b   = (const float*)d_in[19];

    // ---- workspace carve (time-aliased regions, ~195 MB total) -------------
    char* wsb = (char*)d_ws; size_t off = 0;
    auto carve = [&](size_t bytes) -> void* {
        void* p = wsb + off; off = (off + bytes + 255) & ~(size_t)255; return p;
    };
    float*  hf      = (float*) carve((size_t)NM*ND*4);
    __bf16* hb      = (__bf16*)carve((size_t)NM*ND*2);
    float*  out1f   = (float*) carve((size_t)NM*ND*4);
    __bf16* out1b   = (__bf16*)carve((size_t)NM*ND*2);
    float*  projf   = (float*) carve((size_t)NM*ND*4);
    __bf16* wqkvT   = (__bf16*)carve((size_t)3*512*256*2);      // [x][512][256]
    __bf16* wtnsT   = (__bf16*)carve((size_t)3*1024*256*2);     // [x][1024][256]
    __bf16* outwT   = (__bf16*)carve((size_t)256*1536*2);       // [256][1536]
    __bf16* ff1wT   = (__bf16*)carve((size_t)1024*256*2);       // [1024][256]
    __bf16* ff2wT   = (__bf16*)carve((size_t)256*1024*2);       // [256][1024]
    char* R1a = (char*)carve((size_t)NB*NH*SP*SP*2);            // cls_bf -> attn_bf (25.2 MB)
    char* R1b = (char*)carve((size_t)NB*NH*SP*SP*4);            // tns_bf -> scores_f -> ctx_bf (50.3 MB)
    __bf16* qkv   = (__bf16*)carve((size_t)3*NB*NH*SP*ND*2);    // 72 MB
    __bf16* ffact = (__bf16*)carve((size_t)NM*NFF*2);

    __bf16* clsb   = (__bf16*)R1a;  __bf16* attnb = (__bf16*)R1a;
    __bf16* tnsb   = (__bf16*)R1b;  float* scoresf = (float*)R1b; __bf16* ctxb = (__bf16*)R1b;
    const long hslab = (long)NB*NH*SP*ND;                       // 12,582,912
    __bf16* qq = qkv; __bf16* kk = qkv + hslab; __bf16* vvT = qkv + 2*hslab;

    // ---- embedding ---------------------------------------------------------
    embed_kernel<<<(NM*ND)/256, 256, 0, stream>>>(tokens, tok_emb, pos_emb, hf, hb);

    // ---- layers ------------------------------------------------------------
    for (int l = 0; l < NL; ++l) {
        // transpose-convert layer weights to bf16 [N][K]
        f32_to_bf16_t<<<dim3(512,1,3),  256, 0, stream>>>(wqkv_w + (long)l*393216, wqkvT, 256, 512);
        f32_to_bf16_t<<<dim3(1536,1,1), 256, 0, stream>>>(out_w  + (long)l*393216, outwT, 1536, 256);
        f32_to_bf16_t<<<dim3(1024,1,1), 256, 0, stream>>>(ff1_w  + (long)l*262144, ff1wT, 256, 1024);
        f32_to_bf16_t<<<dim3(1024,1,1), 256, 0, stream>>>(ff2_w  + (long)l*262144, ff2wT, 1024, 256);
        build_wtns<<<3072, 256, 0, stream>>>(A1 + (long)l*96, A2 + (long)l*384,
                                             A3 + (long)l*384, A4 + (long)l*96, wtnsT);

        // classical q/k/v: [3] x (6400x512x256)
        gemm_bf16_wmma<<<dim3(8,100,3), 256, 0, stream>>>(
            hb, wqkvT, wqkv_b + (long)l*1536, nullptr, clsb,
            NM, 256, 256, 256, 512,
            1, 0,0, 131072,0, (long)NM*512,0, 512, 0);

        // tensor-network q/k/v: [3] x (6400x1024x256)
        gemm_bf16_wmma<<<dim3(16,100,3), 256, 0, stream>>>(
            hb, wtnsT, tnb + (long)l*3072, nullptr, tnsb,
            NM, 256, 256, 256, 1024,
            1, 0,0, 262144,0, (long)NM*1024,0, 1024, 0);

        assemble_qkv<<<(int)((3L*NB*NH*SP*ND)/256), 256, 0, stream>>>(clsb, tnsb, qkv);

        // scores = Q * K^T per (b,h): [192] x (256x256x256), padded rows garbage
        gemm_bf16_wmma<<<dim3(4,4,NB*NH), 256, 0, stream>>>(
            qq, kk, nullptr, scoresf, nullptr,
            SP, 256, 256, 256, SP,
            1, (long)SP*ND,0, (long)SP*ND,0, (long)SP*SP,0, 0, 0);

        softmax_bias<<<dim3(NS, NB*NH), SP, 0, stream>>>(scoresf, tokens, attnb);

        // ctx = P * V per (b,h): B = V^T [256][SP]; store rows < 200 into [B,S,H*D]
        gemm_bf16_wmma<<<dim3(4,4,NB*NH), 256, 0, stream>>>(
            attnb, vvT, nullptr, nullptr, ctxb,
            NS, SP, SP, SP, NH*ND,
            NH, (long)NH*SP*SP,(long)SP*SP, (long)NH*SP*ND,(long)SP*ND,
            (long)NS*NH*ND,(long)ND, 0, 0);

        // out projection: 6400x256x1536
        gemm_bf16_wmma<<<dim3(4,100,1), 256, 0, stream>>>(
            ctxb, outwT, out_b + (long)l*256, projf, nullptr,
            NM, NH*ND, NH*ND, NH*ND, ND,
            1, 0,0, 0,0, 0,0, 0, 0);

        add_layernorm<<<NM, 256, 0, stream>>>(hf, projf,
            ln1_g + (long)l*ND, ln1_b + (long)l*ND, out1f, out1b);

        // ffn1 + relu: 6400x1024x256
        gemm_bf16_wmma<<<dim3(16,100,1), 256, 0, stream>>>(
            out1b, ff1wT, ff1_b + (long)l*NFF, nullptr, ffact,
            NM, 256, 256, 256, NFF,
            1, 0,0, 0,0, 0,0, 0, 1);

        // ffn2: 6400x256x1024
        gemm_bf16_wmma<<<dim3(4,100,1), 256, 0, stream>>>(
            ffact, ff2wT, ff2_b + (long)l*ND, projf, nullptr,
            NM, NFF, NFF, NFF, ND,
            1, 0,0, 0,0, 0,0, 0, 0);

        add_layernorm<<<NM, 256, 0, stream>>>(out1f, projf,
            ln2_g + (long)l*ND, ln2_b + (long)l*ND, hf, hb);
    }

    copy_f32<<<(NM*ND)/256, 256, 0, stream>>>(hf, (float*)d_out, (long)NM*ND);
}
